// EnhancedGNN_77747497992436
// MI455X (gfx1250) — matrix operand order repaired and checked by reading, not compiled, not run on hardware
//
#include <hip/hip_runtime.h>
#include <hip/hip_bf16.h>
#include <math.h>

// ---------------- problem constants (match reference) ----------------
#define NN      50000
#define EE      600000
#define EA      (EE + NN)     // edges + self loops
#define BB      64
#define F_IN    4
#define HH      128
#define HEADS   4
#define OUTD    32
#define CC      8
#define LN_EPS  1e-5f

typedef __attribute__((ext_vector_type(16))) _Float16 v16h;
typedef __attribute__((ext_vector_type(8)))  _Float16 v8h;
typedef __attribute__((ext_vector_type(8)))  float    v8f;

// ordered-int encoding for float atomic max (monotone bijection)
__device__ __forceinline__ int enc_f(float f) {
    int i = __float_as_int(f);
    return i >= 0 ? i : (i ^ 0x7fffffff);
}
__device__ __forceinline__ float dec_f(int i) {
    return __int_as_float(i >= 0 ? i : (i ^ 0x7fffffff));
}
#define ENC_NEG_INF ((int)0x807fffff)   // enc_f(-inf)

__device__ __forceinline__ float gelu_exact(float x) {
    return 0.5f * x * (1.0f + erff(x * 0.7071067811865475f));
}

// block-wide sum reduction, blockDim.x == 128
__device__ __forceinline__ float block_sum_128(float v, float* red) {
    int t = threadIdx.x;
    red[t] = v; __syncthreads();
    #pragma unroll
    for (int s = 64; s > 0; s >>= 1) {
        if (t < s) red[t] += red[t + s];
        __syncthreads();
    }
    float r = red[0]; __syncthreads();
    return r;
}

// ---------------- node encoder: GELU(LN(x @ enc_w + enc_b)) ----------------
__global__ void enc_kernel(const float* __restrict__ x,
                           const float* __restrict__ ew, const float* __restrict__ eb,
                           const float* __restrict__ g,  const float* __restrict__ b,
                           float* __restrict__ h, _Float16* __restrict__ hf16) {
    __shared__ float red[128];
    int n = blockIdx.x, f = threadIdx.x;
    float v = eb[f];
    #pragma unroll
    for (int i = 0; i < F_IN; ++i) v += x[(size_t)n * F_IN + i] * ew[i * HH + f];
    float mu  = block_sum_128(v, red) * (1.0f / HH);
    float d   = v - mu;
    float var = block_sum_128(d * d, red) * (1.0f / HH);
    float y   = gelu_exact(d * rsqrtf(var + LN_EPS) * g[f] + b[f]);
    h[(size_t)n * HH + f]    = y;
    hf16[(size_t)n * HH + f] = (_Float16)y;
}

// ---------------- convert/transpose W (f32 [k][n]) -> Wt (f16 [n][k]) ----------------
__global__ void convw_kernel(const float* __restrict__ W, _Float16* __restrict__ Wt) {
    int i = blockIdx.x * blockDim.x + threadIdx.x;
    if (i < HH * HH) {
        int k = i >> 7, n = i & 127;
        Wt[n * HH + k] = (_Float16)W[k * HH + n];
    }
}

// ---------------- WMMA GEMM: hW[N,128] = hf16[N,128] @ W[128,128] ----------------
// grid = N/16 blocks, 128 threads (4 waves); wave w owns column tiles 2w, 2w+1.
__global__ void gemm_wmma_kernel(const _Float16* __restrict__ A,   // [N][128] row major f16
                                 const _Float16* __restrict__ Bt,  // [n][k] (W transposed) f16
                                 float* __restrict__ C) {          // [N][128] f32
    int lane = threadIdx.x & 31;
    int wave = threadIdx.x >> 5;
    int r0   = blockIdx.x * 16;
    int m    = lane & 15;
    int khi  = lane >> 4;

    // A fragments for K = 0..127 (4 fragments of 16x32), in this code's contiguous k order:
    // lane(m, khi): VGPR0..7 hold K = khi*16 + 0..15 (contiguous, matching B)
    union { v16h v; v8h h[2]; } afr[4];
    const _Float16* arow = A + (size_t)(r0 + m) * HH + khi * 16;
    #pragma unroll
    for (int kt = 0; kt < 4; ++kt) {
        afr[kt].h[0] = *(const v8h*)(arow + kt * 32);
        afr[kt].h[1] = *(const v8h*)(arow + kt * 32 + 8);
    }

    #pragma unroll
    for (int cti = 0; cti < 2; ++cti) {
        int ct = wave * 2 + cti;
        // B layout: lane(n=lane&15, khi): VGPR0..7 hold K = khi*16 + 0..15 (contiguous in Wt[n][k])
        const _Float16* bcol = Bt + (size_t)(ct * 16 + (lane & 15)) * HH + khi * 16;
        v8f acc = {};
        #pragma unroll
        for (int kt = 0; kt < 4; ++kt) {
            v16h bv = *(const v16h*)(bcol + kt * 32);
            acc = __builtin_amdgcn_wmma_f32_16x16x32_f16(
                false, afr[kt].v, false, bv, (short)0, acc, false, false);
        }
        // C/D layout: VGPR r holds (M = r + (lane>=16 ? 8:0), N = lane&15)
        int rofs = khi * 8;
        int col  = ct * 16 + (lane & 15);
        #pragma unroll
        for (int r = 0; r < 8; ++r)
            C[(size_t)(r0 + rofs + r) * HH + col] = acc[r];
    }
}

// ---------------- per-node attention coefficients ----------------
__global__ void att_node_kernel(const float* __restrict__ hW,
                                const float* __restrict__ as, const float* __restrict__ ad,
                                float* __restrict__ a_src, float* __restrict__ a_dst) {
    int i = blockIdx.x * blockDim.x + threadIdx.x;
    if (i >= NN * HEADS) return;
    int n = i >> 2, hh = i & 3;
    const float* row = hW + (size_t)n * HH + hh * OUTD;
    float s1 = 0.f, s2 = 0.f;
    #pragma unroll
    for (int d = 0; d < OUTD; ++d) { s1 += row[d] * as[hh * OUTD + d]; s2 += row[d] * ad[hh * OUTD + d]; }
    a_src[i] = s1; a_dst[i] = s2;
}

// ---------------- per-layer accumulator init ----------------
__global__ void init_layer_kernel(float* __restrict__ agg, int* __restrict__ mraw,
                                  float* __restrict__ denom) {
    int i = blockIdx.x * blockDim.x + threadIdx.x;
    if (i < NN * HH) agg[i] = 0.f;
    if (i < NN * HEADS) { mraw[i] = ENC_NEG_INF; denom[i] = 0.f; }
}

// ---------------- edge pass 1: leaky-relu logits + segment max ----------------
__global__ void edge_logit_kernel(const int* __restrict__ ei,
                                  const float* __restrict__ a_src, const float* __restrict__ a_dst,
                                  float* __restrict__ elog, int* __restrict__ mraw) {
    int e = blockIdx.x * blockDim.x + threadIdx.x;
    if (e >= EA) return;
    int s, d;
    if (e < EE) { s = ei[e]; d = ei[EE + e]; } else { s = d = e - EE; }
    #pragma unroll
    for (int hh = 0; hh < HEADS; ++hh) {
        float v = a_src[s * HEADS + hh] + a_dst[d * HEADS + hh];
        v = v > 0.f ? v : 0.2f * v;
        elog[(size_t)e * HEADS + hh] = v;
        atomicMax(&mraw[d * HEADS + hh], enc_f(v));
    }
}

// ---------------- edge pass 2: exp(l - m) + segment sum ----------------
__global__ void edge_exp_kernel(const int* __restrict__ ei, float* __restrict__ elog,
                                const int* __restrict__ mraw, float* __restrict__ denom) {
    int e = blockIdx.x * blockDim.x + threadIdx.x;
    if (e >= EA) return;
    int d = (e < EE) ? ei[EE + e] : (e - EE);
    #pragma unroll
    for (int hh = 0; hh < HEADS; ++hh) {
        float m  = dec_f(mraw[d * HEADS + hh]);
        float ex = expf(elog[(size_t)e * HEADS + hh] - m);
        elog[(size_t)e * HEADS + hh] = ex;
        atomicAdd(&denom[d * HEADS + hh], ex);
    }
}

// ---------------- edge pass 3: agg[dst] += hW[src] * alpha ----------------
__global__ void edge_aggr_kernel(const int* __restrict__ ei, const float* __restrict__ elog,
                                 const float* __restrict__ denom, const float* __restrict__ hW,
                                 float* __restrict__ agg) {
    int e = blockIdx.x, f = threadIdx.x;
    int s, d;
    if (e < EE) { s = ei[e]; d = ei[EE + e]; } else { s = d = e - EE; }
    int hh = f >> 5;
    float alpha = elog[(size_t)e * HEADS + hh] / (denom[d * HEADS + hh] + 1e-16f);
    atomicAdd(&agg[(size_t)d * HH + f], hW[(size_t)s * HH + f] * alpha);
}

// ---------------- node finalize: h += GELU(LN(agg + bias)) ----------------
__global__ void finalize_kernel(const float* __restrict__ agg, const float* __restrict__ bias,
                                const float* __restrict__ g, const float* __restrict__ b,
                                float* __restrict__ h, _Float16* __restrict__ hf16) {
    __shared__ float red[128];
    int n = blockIdx.x, f = threadIdx.x;
    float v   = agg[(size_t)n * HH + f] + bias[f];
    float mu  = block_sum_128(v, red) * (1.0f / HH);
    float d   = v - mu;
    float var = block_sum_128(d * d, red) * (1.0f / HH);
    float y   = gelu_exact(d * rsqrtf(var + LN_EPS) * g[f] + b[f]);
    float hn  = h[(size_t)n * HH + f] + y;
    h[(size_t)n * HH + f]    = hn;
    hf16[(size_t)n * HH + f] = (_Float16)hn;
}

// ---------------- pooling ----------------
__global__ void init_pool_kernel(float* __restrict__ psum, int* __restrict__ pmax,
                                 float* __restrict__ pcnt) {
    int i = blockIdx.x * blockDim.x + threadIdx.x;
    if (i < BB * HH) { psum[i] = 0.f; pmax[i] = ENC_NEG_INF; }
    if (i < BB) pcnt[i] = 0.f;
}

__global__ void pool_kernel(const float* __restrict__ h, const int* __restrict__ batch,
                            float* __restrict__ psum, int* __restrict__ pmax,
                            float* __restrict__ pcnt) {
    int n = blockIdx.x, f = threadIdx.x;
    int b = batch[n];
    float v = h[(size_t)n * HH + f];
    atomicAdd(&psum[b * HH + f], v);
    atomicMax(&pmax[b * HH + f], enc_f(v));
    if (f == 0) atomicAdd(&pcnt[b], 1.0f);
}

// ---------------- per-graph head: ts-MLP + concat + LN + classifier ----------------
__global__ void head_kernel(const float* __restrict__ psum, const int* __restrict__ pmax,
                            const float* __restrict__ pcnt, const float* __restrict__ ts,
                            const float* __restrict__ tw1, const float* __restrict__ tb1,
                            const float* __restrict__ tlg, const float* __restrict__ tlb,
                            const float* __restrict__ tw2, const float* __restrict__ tb2,
                            const float* __restrict__ clg, const float* __restrict__ clb,
                            const float* __restrict__ cw1, const float* __restrict__ cb1,
                            const float* __restrict__ cw2, const float* __restrict__ cb2,
                            float* __restrict__ out) {
    __shared__ float z[320];
    __shared__ float red[128];
    __shared__ float t1[64];
    __shared__ float zz[128];
    int b = blockIdx.x, t = threadIdx.x;

    float cnt = fmaxf(pcnt[b], 1.0f);
    z[t]       = psum[b * HH + t] / cnt;     // mean pool
    z[128 + t] = dec_f(pmax[b * HH + t]);    // max pool

    // trackster MLP layer 1
    if (t < 64) {
        float s = tb1[t];
        #pragma unroll
        for (int i = 0; i < 3; ++i) s += ts[b * 3 + i] * tw1[i * 64 + t];
        t1[t] = s;
    }
    __syncthreads();
    float v64 = (t < 64) ? t1[t] : 0.f;
    float mu  = block_sum_128(v64, red) * (1.0f / 64.f);
    float dv  = (t < 64) ? (t1[t] - mu) : 0.f;
    float var = block_sum_128(dv * dv, red) * (1.0f / 64.f);
    if (t < 64) t1[t] = gelu_exact(dv * rsqrtf(var + LN_EPS) * tlg[t] + tlb[t]);
    __syncthreads();
    if (t < 64) {
        float s = tb2[t];
        for (int j = 0; j < 64; ++j) s += t1[j] * tw2[j * 64 + t];
        z[256 + t] = s;
    }
    __syncthreads();

    // LayerNorm over 320-wide concat
    float acc = z[t] + z[t + 128] + ((t < 64) ? z[t + 256] : 0.f);
    mu = block_sum_128(acc, red) * (1.0f / 320.f);
    float d0 = z[t] - mu, d1 = z[t + 128] - mu, d2 = (t < 64) ? (z[t + 256] - mu) : 0.f;
    var = block_sum_128(d0 * d0 + d1 * d1 + d2 * d2, red) * (1.0f / 320.f);
    float rs = rsqrtf(var + LN_EPS);
    z[t]       = d0 * rs * clg[t]       + clb[t];
    z[t + 128] = d1 * rs * clg[t + 128] + clb[t + 128];
    if (t < 64) z[t + 256] = d2 * rs * clg[t + 256] + clb[t + 256];
    __syncthreads();

    // classifier layer 1 (320 -> 128) + GELU
    {
        float s = cb1[t];
        for (int i = 0; i < 320; ++i) s += z[i] * cw1[i * HH + t];
        zz[t] = gelu_exact(s);
    }
    __syncthreads();
    // classifier layer 2 (128 -> 8)
    if (t < CC) {
        float s = cb2[t];
        for (int j = 0; j < HH; ++j) s += zz[j] * cw2[j * CC + t];
        out[b * CC + t] = s;
    }
}

// ============================ launcher ============================
extern "C" void kernel_launch(void* const* d_in, const int* in_sizes, int n_in,
                              void* d_out, int out_size, void* d_ws, size_t ws_size,
                              hipStream_t stream) {
    // inputs in setup_inputs() dict order (params flattened in insertion order)
    const float* x      = (const float*)d_in[0];
    const int*   ei     = (const int*)  d_in[1];
    const int*   batch  = (const int*)  d_in[2];
    const float* ts     = (const float*)d_in[3];
    const float* enc_w  = (const float*)d_in[4];
    const float* enc_b  = (const float*)d_in[5];
    const float* enc_g  = (const float*)d_in[6];
    const float* enc_bn = (const float*)d_in[7];
    // layers: W, att_src, att_dst, bias, ln_g, ln_b  (x3) at indices 8..25
    const float* tw1 = (const float*)d_in[26];
    const float* tb1 = (const float*)d_in[27];
    const float* tlg = (const float*)d_in[28];
    const float* tlb = (const float*)d_in[29];
    const float* tw2 = (const float*)d_in[30];
    const float* tb2 = (const float*)d_in[31];
    const float* clg = (const float*)d_in[32];
    const float* clb = (const float*)d_in[33];
    const float* cw1 = (const float*)d_in[34];
    const float* cb1 = (const float*)d_in[35];
    const float* cw2 = (const float*)d_in[36];
    const float* cb2 = (const float*)d_in[37];
    float* out = (float*)d_out;

    // workspace layout (all offsets 256B-aligned)
    char* w = (char*)d_ws;
    float*    h     = (float*)w;     w += (size_t)NN * HH * sizeof(float);     // 25.6 MB
    float*    hW    = (float*)w;     w += (size_t)NN * HH * sizeof(float);     // 25.6 MB
    float*    agg   = (float*)w;     w += (size_t)NN * HH * sizeof(float);     // 25.6 MB
    _Float16* hf16  = (_Float16*)w;  w += (size_t)NN * HH * sizeof(_Float16);  // 12.8 MB
    _Float16* Wt    = (_Float16*)w;  w += (size_t)HH * HH * sizeof(_Float16);  // 32 KB
    float*    a_src = (float*)w;     w += (size_t)NN * HEADS * sizeof(float);
    float*    a_dst = (float*)w;     w += (size_t)NN * HEADS * sizeof(float);
    int*      mraw  = (int*)w;       w += (size_t)NN * HEADS * sizeof(int);
    float*    denom = (float*)w;     w += (size_t)NN * HEADS * sizeof(float);
    float*    elog  = (float*)w;     w += (size_t)EA * HEADS * sizeof(float);  // 10.4 MB
    float*    psum  = (float*)w;     w += (size_t)BB * HH * sizeof(float);
    int*      pmax  = (int*)w;       w += (size_t)BB * HH * sizeof(int);
    float*    pcnt  = (float*)w;     w += 256;

    // 1) encoder
    enc_kernel<<<NN, 128, 0, stream>>>(x, enc_w, enc_b, enc_g, enc_bn, h, hf16);

    // 2) three GAT layers
    for (int l = 0; l < 3; ++l) {
        const float* Wl   = (const float*)d_in[8 + 6 * l + 0];
        const float* asl  = (const float*)d_in[8 + 6 * l + 1];
        const float* adl  = (const float*)d_in[8 + 6 * l + 2];
        const float* bsl  = (const float*)d_in[8 + 6 * l + 3];
        const float* lgl  = (const float*)d_in[8 + 6 * l + 4];
        const float* lbl  = (const float*)d_in[8 + 6 * l + 5];

        convw_kernel<<<(HH * HH + 255) / 256, 256, 0, stream>>>(Wl, Wt);
        gemm_wmma_kernel<<<NN / 16, 128, 0, stream>>>(hf16, Wt, hW);
        att_node_kernel<<<(NN * HEADS + 255) / 256, 256, 0, stream>>>(hW, asl, adl, a_src, a_dst);
        init_layer_kernel<<<(NN * HH + 255) / 256, 256, 0, stream>>>(agg, mraw, denom);
        edge_logit_kernel<<<(EA + 255) / 256, 256, 0, stream>>>(ei, a_src, a_dst, elog, mraw);
        edge_exp_kernel<<<(EA + 255) / 256, 256, 0, stream>>>(ei, elog, mraw, denom);
        edge_aggr_kernel<<<EA, 128, 0, stream>>>(ei, elog, denom, hW, agg);
        finalize_kernel<<<NN, 128, 0, stream>>>(agg, bsl, lgl, lbl, h, hf16);
    }

    // 3) pooling + heads
    init_pool_kernel<<<(BB * HH + 255) / 256, 256, 0, stream>>>(psum, pmax, pcnt);
    pool_kernel<<<NN, 128, 0, stream>>>(h, batch, psum, pmax, pcnt);
    head_kernel<<<BB, 128, 0, stream>>>(psum, pmax, pcnt, ts,
                                        tw1, tb1, tlg, tlb, tw2, tb2,
                                        clg, clb, cw1, cb1, cw2, cb2, out);
}
